// LSTM_27556510171461
// MI455X (gfx1250) — compile-verified, matrix operand order checked by reading, not run on hardware
//
#include <hip/hip_runtime.h>

// ---------------- Types ----------------
typedef unsigned short ushort_t;
typedef __attribute__((ext_vector_type(16))) __bf16 v16bf;
typedef __attribute__((ext_vector_type(8)))  __bf16 v8bf;
typedef __attribute__((ext_vector_type(8)))  float  v8f;

#define B_  64
#define T_  512
#define F_  512
#define H_  1024
#define G4_ 4096      // 4*H
#define NWG 64        // workgroups in persistent kernel (one per 16 hidden cols)
#define NTH 512       // 16 wave32 per workgroup
#define HB_ (B_*H_)   // elements of one h parity buffer

// ---------------- Helpers ----------------
__device__ __forceinline__ ushort_t f2bf(float f) {
    unsigned u = __float_as_uint(f);
    unsigned r = (u + 0x7FFFu + ((u >> 16) & 1u)) >> 16;   // round-to-nearest-even
    return (ushort_t)r;
}

__device__ __forceinline__ float sigf(float x) {
    return 1.f / (1.f + __expf(-x));
}
__device__ __forceinline__ float tanh_fast(float x) {
    x = fminf(10.f, fmaxf(-10.f, x));
    float e = __expf(2.f * x);
    return (e - 1.f) / (e + 1.f);
}

// ---------------- Prep kernels ----------------
__global__ void cvt_f32_bf16(const float* __restrict__ src, ushort_t* __restrict__ dst, int n) {
    int i = blockIdx.x * blockDim.x + threadIdx.x;
    if (i < n) dst[i] = f2bf(src[i]);
}

// x [B,T,F] fp32 -> xbf [T,B,F] bf16
__global__ void x_transpose_bf16(const float* __restrict__ x, ushort_t* __restrict__ xbf) {
    int i = blockIdx.x * blockDim.x + threadIdx.x;   // total B*T*F, i = ((b*T)+t)*F+f
    int f = i & (F_ - 1);
    int r = i >> 9;            // /F_
    int t = r & (T_ - 1);
    int b = r >> 9;            // /T_
    xbf[(size_t)t * (B_ * F_) + b * F_ + f] = f2bf(x[i]);
}

__global__ void bias_sum(const float* bi0, const float* bh0,
                         const float* bi1, const float* bh1,
                         float* bias0, float* bias1) {
    int i = blockIdx.x * blockDim.x + threadIdx.x;
    if (i < G4_) {
        bias0[i] = bi0[i] + bh0[i];
        bias1[i] = bi1[i] + bh1[i];
    }
}

// ---------------- Device-wide barrier (monotonic counter) ----------------
__device__ __forceinline__ void grid_barrier(unsigned* ctr, unsigned gen, int tid) {
    __threadfence();           // release my stores to device scope
    __syncthreads();
    if (tid == 0) {
        __hip_atomic_fetch_add(ctr, 1u, __ATOMIC_RELEASE, __HIP_MEMORY_SCOPE_AGENT);
        const unsigned target = gen * (unsigned)NWG;
        while (__hip_atomic_load(ctr, __ATOMIC_ACQUIRE, __HIP_MEMORY_SCOPE_AGENT) < target) {
            __builtin_amdgcn_s_sleep(1);
        }
    }
    __syncthreads();
}

// ---------------- One LSTM layer step (GEMM via WMMA + cell elementwise) -----
// gates = [Ax | Ah] @ [Wx | Wh]^T + bias ; c,h update on this WG's 16 columns.
__device__ __forceinline__ void lstm_layer_step(
    int tid, int n0,
    const ushort_t* __restrict__ Ax, int Kx,        // [64, Kx] bf16 row-major
    const ushort_t* __restrict__ Ah,                // [64, 1024] bf16
    const ushort_t* __restrict__ Wx,                // [4096, Kx] bf16 (N-major)
    const ushort_t* __restrict__ Wh,                // [4096, 1024] bf16
    const float* __restrict__ bias,                 // [4096]
    float* gbuf,                                    // LDS [64][64] fp32 gates
    float* cbuf,                                    // LDS [64][16] fp32 cell state
    ushort_t* __restrict__ hbf_out,                 // [64, 1024] bf16 (this step's h)
    float* __restrict__ f32a, long sa,              // optional fp32 h sinks
    float* __restrict__ f32b, long sb)
{
    const int lane = tid & 31;
    const int w    = tid >> 5;     // wave 0..15
    const int m    = w & 3;        // M tile (16 rows of batch)
    const int q    = w >> 2;       // gate 0..3 (i,f,g,o)
    const int cc   = lane & 15;
    const int hi   = lane >> 4;    // half-wave select
    const int koff = hi * 8;       // ISA 16-bit A/B lane K-offset
    const int arow = m * 16 + cc;
    const long brow = (long)(q * H_ + n0 + cc);

    const float bv = bias[q * H_ + n0 + cc];
    v8f acc = { bv, bv, bv, bv, bv, bv, bv, bv };

    // K segment 1: x-part
    {
        const ushort_t* aP = Ax + (long)arow * Kx + koff;
        const ushort_t* bP = Wx + brow * Kx + koff;
        #pragma unroll 4
        for (int k0 = 0; k0 < Kx; k0 += 32) {
            v8bf a0 = *(const v8bf*)(aP + k0);
            v8bf a1 = *(const v8bf*)(aP + k0 + 16);
            v8bf b0 = *(const v8bf*)(bP + k0);
            v8bf b1 = *(const v8bf*)(bP + k0 + 16);
            v16bf av = __builtin_shufflevector(a0, a1, 0,1,2,3,4,5,6,7,8,9,10,11,12,13,14,15);
            v16bf bvv = __builtin_shufflevector(b0, b1, 0,1,2,3,4,5,6,7,8,9,10,11,12,13,14,15);
            acc = __builtin_amdgcn_wmma_f32_16x16x32_bf16(false, av, false, bvv,
                                                          (short)0, acc, false, false);
        }
    }
    // K segment 2: recurrent h-part (K = 1024)
    {
        const ushort_t* aP = Ah + (long)arow * H_ + koff;
        const ushort_t* bP = Wh + brow * H_ + koff;
        #pragma unroll 4
        for (int k0 = 0; k0 < H_; k0 += 32) {
            v8bf a0 = *(const v8bf*)(aP + k0);
            v8bf a1 = *(const v8bf*)(aP + k0 + 16);
            v8bf b0 = *(const v8bf*)(bP + k0);
            v8bf b1 = *(const v8bf*)(bP + k0 + 16);
            v16bf av = __builtin_shufflevector(a0, a1, 0,1,2,3,4,5,6,7,8,9,10,11,12,13,14,15);
            v16bf bvv = __builtin_shufflevector(b0, b1, 0,1,2,3,4,5,6,7,8,9,10,11,12,13,14,15);
            acc = __builtin_amdgcn_wmma_f32_16x16x32_bf16(false, av, false, bvv,
                                                          (short)0, acc, false, false);
        }
    }

    // Spill D tile to LDS: C/D layout -> row = m*16 + r + 8*hi, col = q*16 + cc
    #pragma unroll
    for (int r = 0; r < 8; ++r)
        gbuf[(m * 16 + r + 8 * hi) * 64 + q * 16 + cc] = acc[r];
    __syncthreads();

    // Cell elementwise on this WG's 64x16 slice (2 elements / thread)
    #pragma unroll
    for (int p = 0; p < 2; ++p) {
        int e  = tid + p * NTH;    // 0..1023
        int b  = e >> 4;
        int c  = e & 15;
        float gi = gbuf[b * 64 + 0  + c];
        float gf = gbuf[b * 64 + 16 + c];
        float gg = gbuf[b * 64 + 32 + c];
        float go = gbuf[b * 64 + 48 + c];
        float cold = cbuf[b * 16 + c];
        float cn = sigf(gf) * cold + sigf(gi) * tanh_fast(gg);
        float hn = sigf(go) * tanh_fast(cn);
        cbuf[b * 16 + c] = cn;
        hbf_out[(size_t)b * H_ + n0 + c] = f2bf(hn);
        if (f32a) f32a[(long)b * sa + n0 + c] = hn;
        if (f32b) f32b[(long)b * sb + n0 + c] = hn;
    }
    __syncthreads();
}

// ---------------- Persistent recurrent kernel ----------------
__global__ __launch_bounds__(NTH, 1)
void lstm_persistent(const ushort_t* __restrict__ xbf,
                     const ushort_t* __restrict__ Wi0, const ushort_t* __restrict__ Wh0,
                     const ushort_t* __restrict__ Wi1, const ushort_t* __restrict__ Wh1,
                     const float* __restrict__ bias0, const float* __restrict__ bias1,
                     ushort_t* __restrict__ h0bf,     // [2][B*H]
                     ushort_t* __restrict__ h1bf,     // [2][B*H]
                     unsigned* __restrict__ barrier_ctr,
                     float* __restrict__ out)
{
    __shared__ float gbuf[64 * 64];
    __shared__ float c0buf[64 * 16];
    __shared__ float c1buf[64 * 16];

    const int tid = threadIdx.x;
    const int n0  = blockIdx.x * 16;

    for (int i = tid; i < 64 * 16; i += NTH) { c0buf[i] = 0.f; c1buf[i] = 0.f; }
    __syncthreads();

    float* outSeq = out;                                 // [B,T,H]
    float* hT = out + (long)B_ * T_ * H_;                // [2,B,H]
    float* cT = hT + 2L * B_ * H_;                       // [2,B,H]

    // h0[t] lives in h0bf[t&1]; h0[-1]=h0bf[1]=zeros. Same for h1.
    // Layer0 step 0:
    lstm_layer_step(tid, n0,
                    xbf, F_, h0bf + 1 * HB_,
                    Wi0, Wh0, bias0, gbuf, c0buf,
                    h0bf + 0 * HB_,
                    nullptr, 0, nullptr, 0);

    for (int t = 0; t < T_; ++t) {
        grid_barrier(barrier_ctr, (unsigned)(t + 1), tid);

        // Layer1 step t: x-part = h0[t], h-part = h1[t-1]
        {
            float* fa = outSeq + (long)t * H_;  long sa = (long)T_ * H_;
            float* fb = (t == T_ - 1) ? (hT + (long)B_ * H_) : nullptr;
            lstm_layer_step(tid, n0,
                            h0bf + (size_t)(t & 1) * HB_, H_,
                            h1bf + (size_t)((t + 1) & 1) * HB_,
                            Wi1, Wh1, bias1, gbuf, c1buf,
                            h1bf + (size_t)(t & 1) * HB_,
                            fa, sa, fb, (long)H_);
        }
        // Layer0 step t+1 (no extra barrier needed: parity double-buffer)
        if (t + 1 < T_) {
            const int u = t + 1;
            float* fa = (u == T_ - 1) ? hT : nullptr;
            lstm_layer_step(tid, n0,
                            xbf + (size_t)u * (B_ * F_), F_,
                            h0bf + (size_t)((u + 1) & 1) * HB_,
                            Wi0, Wh0, bias0, gbuf, c0buf,
                            h0bf + (size_t)(u & 1) * HB_,
                            fa, (long)H_, nullptr, 0);
        }
    }

    // Final cell states from LDS
    #pragma unroll
    for (int p = 0; p < 2; ++p) {
        int e = tid + p * NTH;
        int b = e >> 4;
        int c = e & 15;
        cT[(long)b * H_ + n0 + c]                 = c0buf[b * 16 + c];
        cT[(long)B_ * H_ + (long)b * H_ + n0 + c] = c1buf[b * 16 + c];
    }
}

// ---------------- Host entry ----------------
extern "C" void kernel_launch(void* const* d_in, const int* in_sizes, int n_in,
                              void* d_out, int out_size, void* d_ws, size_t ws_size,
                              hipStream_t stream) {
    const float* x   = (const float*)d_in[0];
    const float* Wi0 = (const float*)d_in[1];
    const float* Wh0 = (const float*)d_in[2];
    const float* bi0 = (const float*)d_in[3];
    const float* bh0 = (const float*)d_in[4];
    const float* Wi1 = (const float*)d_in[5];
    const float* Wh1 = (const float*)d_in[6];
    const float* bi1 = (const float*)d_in[7];
    const float* bh1 = (const float*)d_in[8];

    char* ws = (char*)d_ws;
    size_t off = 0;
    auto alloc = [&](size_t bytes) -> void* {
        void* p = ws + off;
        off += (bytes + 255) & ~(size_t)255;
        return p;
    };
    unsigned* ctr   = (unsigned*)alloc(256);
    ushort_t* xbf   = (ushort_t*)alloc((size_t)T_ * B_ * F_ * 2);
    ushort_t* wi0b  = (ushort_t*)alloc((size_t)G4_ * F_ * 2);
    ushort_t* wh0b  = (ushort_t*)alloc((size_t)G4_ * H_ * 2);
    ushort_t* wi1b  = (ushort_t*)alloc((size_t)G4_ * H_ * 2);
    ushort_t* wh1b  = (ushort_t*)alloc((size_t)G4_ * H_ * 2);
    float*    bias0 = (float*)alloc((size_t)G4_ * 4);
    float*    bias1 = (float*)alloc((size_t)G4_ * 4);
    ushort_t* h0bf  = (ushort_t*)alloc((size_t)2 * HB_ * 2);
    ushort_t* h1bf  = (ushort_t*)alloc((size_t)2 * HB_ * 2);

    hipMemsetAsync(ctr,  0, 256, stream);
    hipMemsetAsync(h0bf, 0, (size_t)2 * HB_ * 2, stream);
    hipMemsetAsync(h1bf, 0, (size_t)2 * HB_ * 2, stream);

    {
        int n = G4_ * F_;
        cvt_f32_bf16<<<(n + 255) / 256, 256, 0, stream>>>(Wi0, wi0b, n);
    }
    {
        int n = G4_ * H_;
        cvt_f32_bf16<<<(n + 255) / 256, 256, 0, stream>>>(Wh0, wh0b, n);
        cvt_f32_bf16<<<(n + 255) / 256, 256, 0, stream>>>(Wi1, wi1b, n);
        cvt_f32_bf16<<<(n + 255) / 256, 256, 0, stream>>>(Wh1, wh1b, n);
    }
    bias_sum<<<(G4_ + 255) / 256, 256, 0, stream>>>(bi0, bh0, bi1, bh1, bias0, bias1);
    {
        int n = B_ * T_ * F_;
        x_transpose_bf16<<<n / 256, 256, 0, stream>>>(x, xbf);
    }

    lstm_persistent<<<NWG, NTH, 0, stream>>>(
        xbf, wi0b, wh0b, wi1b, wh1b, bias0, bias1,
        h0bf, h1bf, ctr, (float*)d_out);
}
